// TransformerEncoderModel_27771258536729
// MI455X (gfx1250) — matrix operand order repaired and checked by reading, not compiled, not run on hardware
//
#include <hip/hip_runtime.h>
#include <hip/hip_bf16.h>

#define NLY 6
#define DMODEL 256
#define NH 8
#define DKH 32
#define FFD 1024
#define CTXD 64
#define BB 2
#define LL 2048
#define MTOK (BB*LL)

typedef __attribute__((ext_vector_type(16))) _Float16 v16h;
typedef __attribute__((ext_vector_type(8)))  float    v8f;

__device__ __forceinline__ v8f wmma_f16(v16h a, v16h b, v8f c) {
  // v_wmma_f32_16x16x32_f16: D = A(16x32 f16) * B(32x16 f16) + C(16x16 f32)
  return __builtin_amdgcn_wmma_f32_16x16x32_f16(false, a, false, b, (short)0, c, false, false);
}

// A-matrix 16x32 f16 layout (ISA 7.12.2): lane m(0..15): elems0-7 = K0..7, elems8-15 = K16..23
// lane m+16: elems0-7 = K8..15, elems8-15 = K24..31.  rowp = LDS row of 32 f16 (16B aligned).
__device__ __forceinline__ v16h frag_a(const _Float16* rowp, int half) {
  union { v16h v; uint4 u[2]; } t;
  t.u[0] = *(const uint4*)(rowp + half * 8);
  t.u[1] = *(const uint4*)(rowp + 16 + half * 8);
  return t.v;
}
// B-matrix 32x16 f16 layout: lane n(0..15): elems0-7 = K0..7, elems8-15 = K16..23 of col n; lane n+16: K8..15, K24..31.
// rowp = LDS row holding Bt[n][k] (col-major B, 32 f16).
__device__ __forceinline__ v16h frag_b(const _Float16* rowp, int half) {
  union { v16h v; uint4 u[2]; } t;
  t.u[0] = *(const uint4*)(rowp + half * 8);
  t.u[1] = *(const uint4*)(rowp + 16 + half * 8);
  return t.v;
}

__device__ __forceinline__ float gelu_exact(float x) {
  return 0.5f * x * (1.0f + erff(x * 0.70710678118654752f));
}

// wave32 shfl reduction + 8-partial LDS combine (block of 256 threads)
__device__ __forceinline__ float blk_sum256(float v, float* part, int tid) {
  #pragma unroll
  for (int md = 1; md < 32; md <<= 1) v += __shfl_xor(v, md, 32);
  if ((tid & 31) == 0) part[tid >> 5] = v;
  __syncthreads();
  float s = part[0] + part[1] + part[2] + part[3] + part[4] + part[5] + part[6] + part[7];
  __syncthreads();
  return s;
}
__device__ __forceinline__ float blk_max256(float v, float* part, int tid) {
  #pragma unroll
  for (int md = 1; md < 32; md <<= 1) v = fmaxf(v, __shfl_xor(v, md, 32));
  if ((tid & 31) == 0) part[tid >> 5] = v;
  __syncthreads();
  float s = part[0];
  #pragma unroll
  for (int i = 1; i < 8; i++) s = fmaxf(s, part[i]);
  __syncthreads();
  return s;
}

// ---------------- token embedding MLP + RoPE tables ----------------
__global__ __launch_bounds__(256) void k_embed(
    const float* __restrict__ feat, const int* __restrict__ bid,
    const float* __restrict__ emb,
    const float* __restrict__ W1, const float* __restrict__ b1,
    const float* __restrict__ W2, const float* __restrict__ b2,
    float* __restrict__ x, float* __restrict__ cs) {
  int t = blockIdx.x;       // token 0..4095
  int j = threadIdx.x;      // 0..255
  __shared__ float tok[16];
  __shared__ float h[DMODEL];
  if (j < 6) tok[j] = feat[t * 6 + j];
  else if (j < 14) tok[j] = emb[bid[t] * 8 + (j - 6)];
  __syncthreads();
  float acc = b1[j];
  #pragma unroll
  for (int k = 0; k < 14; k++) acc += tok[k] * W1[k * DMODEL + j];
  h[j] = gelu_exact(acc);
  __syncthreads();
  float acc2 = b2[j];
  for (int k = 0; k < DMODEL; k++) acc2 += h[k] * W2[k * DMODEL + j];
  x[(size_t)t * DMODEL + j] = acc2;
  if (j < 16) {
    float pos = feat[t * 6 + 0] * 512.0f;
    float invf = powf(10000.0f, -((float)(2 * j)) / 32.0f);
    float fr = pos * invf;
    cs[(size_t)t * 32 + 2 * j + 0] = cosf(fr);
    cs[(size_t)t * 32 + 2 * j + 1] = sinf(fr);
  }
}

// ---------------- LayerNorm (one block per token, wave32 reductions) ----------------
__global__ __launch_bounds__(256) void k_ln(
    const float* __restrict__ x, const float* __restrict__ g,
    const float* __restrict__ b, float* __restrict__ y) {
  int t = blockIdx.x; int j = threadIdx.x;
  __shared__ float part[8];
  float v = x[(size_t)t * DMODEL + j];
  float mu = blk_sum256(v, part, j) * (1.0f / DMODEL);
  float d = v - mu;
  float var = blk_sum256(d * d, part, j) * (1.0f / DMODEL);
  y[(size_t)t * DMODEL + j] = d * rsqrtf(var + 1e-5f) * g[j] + b[j];
}

// ---------------- WMMA GEMM:  D[M,N] = op(A[M,K] @ W[K,N] + bias [+resid]) ----------------
// block = 256 thr = 8 waves; block tile 128x64; wave tile 32x32 (2x2 wmma)
// Double-buffered LDS: next chunk's global loads issued before WMMAs.
// MODE: 0 = bias, 1 = gelu(bias+), 2 = bias + residual
template<int MODE>
__global__ __launch_bounds__(256) void k_gemm(
    const float* __restrict__ A, const float* __restrict__ W,
    const float* __restrict__ bias, const float* __restrict__ resid,
    float* __restrict__ Dout, int M, int N, int K) {
  const int tid = threadIdx.x;
  const int lane = tid & 31;
  const int w = tid >> 5;
  const int wm = w & 3, wn = w >> 2;
  const int half = lane >> 4, l16 = lane & 15;
  const int m0 = blockIdx.y * 128;
  const int n0 = blockIdx.x * 64;
  __shared__ __align__(16) _Float16 As[2][128][40];
  __shared__ __align__(16) _Float16 Bs[2][64][40];

  // staging assignment: A: thread -> (row ar, 16 k's at ac4); B: (4 n's at bn4, k = bc & bc+16)
  const int ar = tid >> 1;
  const int ac4 = (tid & 1) * 16;
  const int bn4 = (tid & 15) * 4;
  const int bc = tid >> 4;

  float4 areg[4];
  float4 breg[2];
  auto ldAB = [&](int k0) {
    const float* ap = A + (size_t)(m0 + ar) * K + k0 + ac4;
    #pragma unroll
    for (int jj = 0; jj < 4; jj++) areg[jj] = *(const float4*)(ap + 4 * jj);
    breg[0] = *(const float4*)(W + (size_t)(k0 + bc) * N + n0 + bn4);
    breg[1] = *(const float4*)(W + (size_t)(k0 + bc + 16) * N + n0 + bn4);
  };
  auto stAB = [&](int buf) {
    union { _Float16 hh[16]; uint4 u[2]; } p;
    #pragma unroll
    for (int jj = 0; jj < 4; jj++)
      #pragma unroll
      for (int e = 0; e < 4; e++) p.hh[4 * jj + e] = (_Float16)((const float*)&areg[jj])[e];
    *(uint4*)&As[buf][ar][ac4]     = p.u[0];
    *(uint4*)&As[buf][ar][ac4 + 8] = p.u[1];
    #pragma unroll
    for (int e = 0; e < 4; e++) {
      Bs[buf][bn4 + e][bc]      = (_Float16)((const float*)&breg[0])[e];
      Bs[buf][bn4 + e][bc + 16] = (_Float16)((const float*)&breg[1])[e];
    }
  };

  ldAB(0);
  stAB(0);
  __syncthreads();

  v8f c[2][2] = {};
  const int nch = K >> 5;
  for (int ch = 0; ch < nch; ch++) {
    const int cur = ch & 1;
    const bool more = (ch + 1 < nch);
    if (more) ldAB((ch + 1) * 32);          // global loads in flight during WMMAs
    v16h a0 = frag_a(&As[cur][wm * 32 + l16][0], half);
    v16h a1 = frag_a(&As[cur][wm * 32 + 16 + l16][0], half);
    v16h b0 = frag_b(&Bs[cur][wn * 32 + l16][0], half);
    v16h b1 = frag_b(&Bs[cur][wn * 32 + 16 + l16][0], half);
    c[0][0] = wmma_f16(a0, b0, c[0][0]);
    c[0][1] = wmma_f16(a0, b1, c[0][1]);
    c[1][0] = wmma_f16(a1, b0, c[1][0]);
    c[1][1] = wmma_f16(a1, b1, c[1][1]);
    if (more) stAB(1 - cur);
    __syncthreads();
  }

  #pragma unroll
  for (int mt = 0; mt < 2; mt++)
    #pragma unroll
    for (int nt = 0; nt < 2; nt++) {
      const int col = n0 + wn * 32 + nt * 16 + l16;
      const float bv = bias[col];
      const int rbase = m0 + wm * 32 + mt * 16 + 8 * half;
      #pragma unroll
      for (int r = 0; r < 8; r++) {
        const size_t off = (size_t)(rbase + r) * N + col;
        float v = c[mt][nt][r] + bv;
        if (MODE == 1) v = gelu_exact(v);
        if (MODE == 2) v += resid[off];
        Dout[off] = v;
      }
    }
}

// ---------------- Flash attention per (b,h); one block covers 128 query rows ----------------
// Double-buffered K/V chunks of 64 keys; RoPE folded into staging; 1/sqrt(dk) folded into Q.
__global__ __launch_bounds__(256) void k_attn(
    const float* __restrict__ qkv, const float* __restrict__ cs,
    const unsigned char* __restrict__ mask, float* __restrict__ outp) {
  const int bh = blockIdx.x;            // 0..15
  const int b = bh >> 3, h = bh & 7;
  const int q0 = blockIdx.y * 128;
  const int tid = threadIdx.x, lane = tid & 31, w = tid >> 5;
  const int half = lane >> 4, l16 = lane & 15;
  __shared__ __align__(16) _Float16 Qs[8][16][40];
  __shared__ __align__(16) _Float16 Ks[2][64][40];
  __shared__ __align__(16) _Float16 Vs[2][32][80];
  __shared__ __align__(16) _Float16 Ps[8][16][80];
  const float scale = 0.17677669529663687f;   // 32^-0.5, folded into Q

  // K/V staging assignment: key = tid&63, 8 dims starting at d0
  const int key = tid & 63;
  const int d0 = (tid >> 6) * 8;
  float kreg[8], vreg[8];
  auto ldKV = [&](int kc) {
    const int krow = kc + key;
    const float* kp = qkv + ((size_t)(b * LL + krow) * 3 + 1) * DMODEL + h * DKH;
    const float* vp = qkv + ((size_t)(b * LL + krow) * 3 + 2) * DMODEL + h * DKH;
    const float* cp = cs + (size_t)(b * LL + krow) * 32;
    #pragma unroll
    for (int dd = 0; dd < 8; dd++) {
      const int d = d0 + dd;
      if (d < 16) kreg[dd] = kp[d] * cp[2 * d] - kp[d + 16] * cp[2 * d + 1];
      else { const int dl = d - 16; kreg[dd] = kp[dl] * cp[2 * dl + 1] + kp[d] * cp[2 * dl]; }
      vreg[dd] = vp[d];
    }
  };
  auto stKV = [&](int buf) {
    #pragma unroll
    for (int dd = 0; dd < 8; dd++) {
      Ks[buf][key][d0 + dd] = (_Float16)kreg[dd];
      Vs[buf][d0 + dd][key] = (_Float16)vreg[dd];
    }
  };

  // stage this wave's 16 query rows with RoPE (lanes 0..15), plus chunk 0 of K/V
  if (lane < 16) {
    const int m = lane;
    const int qrow = q0 + w * 16 + m;
    const float* qp = qkv + ((size_t)(b * LL + qrow) * 3 + 0) * DMODEL + h * DKH;
    const float* cp = cs + (size_t)(b * LL + qrow) * 32;
    #pragma unroll
    for (int d = 0; d < 16; d++) {
      float c = cp[2 * d], s = cp[2 * d + 1];
      float x1 = qp[d], x2 = qp[d + 16];
      Qs[w][m][d]      = (_Float16)((x1 * c - x2 * s) * scale);
      Qs[w][m][d + 16] = (_Float16)((x1 * s + x2 * c) * scale);
    }
  }
  ldKV(0);
  stKV(0);
  __syncthreads();
  v16h aq = frag_a(&Qs[w][l16][0], half);

  v8f o0 = {}, o1 = {};
  float mrun[8], lrun[8];
  #pragma unroll
  for (int r = 0; r < 8; r++) { mrun[r] = -3.0e38f; lrun[r] = 0.0f; }

  for (int kc = 0; kc < LL; kc += 64) {
    const int cur = (kc >> 6) & 1;
    const bool more = (kc + 64 < LL);
    // branchless mask terms for this lane's 4 columns (loads overlap the WMMAs below)
    float madd[4];
    #pragma unroll
    for (int t = 0; t < 4; t++)
      madd[t] = mask[b * LL + kc + t * 16 + l16] ? 0.0f : -1.0e30f;
    if (more) ldKV(kc + 64);            // next chunk's loads in flight during compute
    // S = Q * K^T  (4 wmma, 16x64 scores in C-layout)
    v8f s[4];
    #pragma unroll
    for (int t = 0; t < 4; t++) {
      v16h bk = frag_b(&Ks[cur][t * 16 + l16][0], half);
      v8f z = {};
      s[t] = wmma_f16(aq, bk, z);
    }
    #pragma unroll
    for (int t = 0; t < 4; t++)
      #pragma unroll
      for (int r = 0; r < 8; r++) s[t][r] += madd[t];
    // online softmax (row = r + 8*half lives across the 16 lanes of this half)
    #pragma unroll
    for (int r = 0; r < 8; r++) {
      float rm = fmaxf(fmaxf(s[0][r], s[1][r]), fmaxf(s[2][r], s[3][r]));
      #pragma unroll
      for (int md = 1; md < 16; md <<= 1) rm = fmaxf(rm, __shfl_xor(rm, md, 32));
      float mn = fmaxf(mrun[r], rm);
      float alpha = __expf(mrun[r] - mn);
      mrun[r] = mn;
      float rs = 0.0f;
      #pragma unroll
      for (int t = 0; t < 4; t++) { float p = __expf(s[t][r] - mn); s[t][r] = p; rs += p; }
      #pragma unroll
      for (int md = 1; md < 16; md <<= 1) rs += __shfl_xor(rs, md, 32);
      lrun[r] = lrun[r] * alpha + rs;
      o0[r] *= alpha; o1[r] *= alpha;
    }
    // spill P (C-layout) to LDS, reload as A-fragments (wave-local round trip)
    #pragma unroll
    for (int t = 0; t < 4; t++)
      #pragma unroll
      for (int r = 0; r < 8; r++)
        Ps[w][r + 8 * half][t * 16 + l16] = (_Float16)s[t][r];
    asm volatile("s_wait_dscnt 0" ::: "memory");
    #pragma unroll
    for (int kb = 0; kb < 2; kb++) {
      v16h ap  = frag_a(&Ps[w][l16][0] + kb * 32, half);
      v16h bv0 = frag_b(&Vs[cur][l16][0] + kb * 32, half);
      v16h bv1 = frag_b(&Vs[cur][16 + l16][0] + kb * 32, half);
      o0 = wmma_f16(ap, bv0, o0);
      o1 = wmma_f16(ap, bv1, o1);
    }
    if (more) stKV(1 - cur);
    __syncthreads();
  }
  // finalize: o -> [b][l][h][dk]  ((h,dk) contiguous == D)
  #pragma unroll
  for (int r = 0; r < 8; r++) {
    const int qrow = q0 + w * 16 + r + 8 * half;
    const float inv = 1.0f / lrun[r];
    const size_t base = ((size_t)(b * LL + qrow) * NH + h) * DKH;
    outp[base + l16]      = o0[r] * inv;
    outp[base + 16 + l16] = o1[r] * inv;
  }
}

// ---------------- pooling: q projection (tiny) ----------------
__global__ __launch_bounds__(256) void k_qproj(
    const float* __restrict__ pq, const float* __restrict__ Wq,
    const float* __restrict__ bq, float* __restrict__ qvec) {
  int j = threadIdx.x;
  __shared__ float sq[DMODEL];
  sq[j] = pq[j];
  __syncthreads();
  float acc = bq[j];
  for (int k = 0; k < DMODEL; k++) acc += sq[k] * Wq[k * DMODEL + j];
  qvec[j] = acc;
}

// ---------------- pooling attention: 1 query vs L keys, per (b,h) ----------------
__global__ __launch_bounds__(256) void k_pool_attn(
    const float* __restrict__ qvec, const float* __restrict__ kh,
    const float* __restrict__ vh, const unsigned char* __restrict__ mask,
    float* __restrict__ po) {
  const int bh = blockIdx.x; const int b = bh >> 3, h = bh & 7;
  const int tid = threadIdx.x;
  const float scale = 0.17677669529663687f;
  __shared__ float qh[DKH];
  __shared__ float part[8];
  __shared__ float ov[DKH];
  if (tid < DKH) { qh[tid] = qvec[h * DKH + tid]; ov[tid] = 0.0f; }
  __syncthreads();
  float lm = -3.0e38f;
  for (int l = tid; l < LL; l += 256) {
    const float* kp = kh + (size_t)(b * LL + l) * DMODEL + h * DKH;
    float s = 0.0f;
    #pragma unroll
    for (int d = 0; d < DKH; d++) s += qh[d] * kp[d];
    s = s * scale + (mask[b * LL + l] ? 0.0f : -1.0e30f);
    lm = fmaxf(lm, s);
  }
  float m = blk_max256(lm, part, tid);
  float ls = 0.0f;
  float lo[DKH];
  #pragma unroll
  for (int d = 0; d < DKH; d++) lo[d] = 0.0f;
  for (int l = tid; l < LL; l += 256) {
    const float* kp = kh + (size_t)(b * LL + l) * DMODEL + h * DKH;
    const float* vp = vh + (size_t)(b * LL + l) * DMODEL + h * DKH;
    float s = 0.0f;
    #pragma unroll
    for (int d = 0; d < DKH; d++) s += qh[d] * kp[d];
    s = s * scale + (mask[b * LL + l] ? 0.0f : -1.0e30f);
    float p = __expf(s - m);
    ls += p;
    #pragma unroll
    for (int d = 0; d < DKH; d++) lo[d] += p * vp[d];
  }
  float denom = blk_sum256(ls, part, tid);
  #pragma unroll
  for (int d = 0; d < DKH; d++) atomicAdd(&ov[d], lo[d]);
  __syncthreads();
  if (tid < DKH) po[b * DMODEL + h * DKH + tid] = ov[tid] / denom;
}

// ---------------- final: out-proj + LN + LN + ctx head, one block per batch ----------------
__global__ __launch_bounds__(256) void k_pool_final(
    const float* __restrict__ po, const float* __restrict__ Wo, const float* __restrict__ bo,
    const float* __restrict__ plg, const float* __restrict__ plb,
    const float* __restrict__ clg, const float* __restrict__ clb,
    const float* __restrict__ ctxW, const float* __restrict__ ctxb,
    float* __restrict__ out) {
  const int b = blockIdx.x; const int j = threadIdx.x;
  __shared__ float sv[DMODEL];
  __shared__ float part[8];
  sv[j] = po[b * DMODEL + j];
  __syncthreads();
  float u = bo[j];
  for (int k = 0; k < DMODEL; k++) u += sv[k] * Wo[k * DMODEL + j];
  __syncthreads();
  // LN 1
  float mu = blk_sum256(u, part, j) * (1.0f / DMODEL);
  float d = u - mu;
  float var = blk_sum256(d * d, part, j) * (1.0f / DMODEL);
  float p1 = d * rsqrtf(var + 1e-5f) * plg[j] + plb[j];
  // LN 2
  float mu2 = blk_sum256(p1, part, j) * (1.0f / DMODEL);
  float d2 = p1 - mu2;
  float var2 = blk_sum256(d2 * d2, part, j) * (1.0f / DMODEL);
  float z = d2 * rsqrtf(var2 + 1e-5f) * clg[j] + clb[j];
  sv[j] = z;
  __syncthreads();
  if (j < CTXD) {
    float acc = ctxb[j];
    for (int k = 0; k < DMODEL; k++) acc += sv[k] * ctxW[k * CTXD + j];
    out[b * CTXD + j] = acc;
  }
}

extern "C" void kernel_launch(void* const* d_in, const int* in_sizes, int n_in,
                              void* d_out, int out_size, void* d_ws, size_t ws_size,
                              hipStream_t stream) {
  const float* feat   = (const float*)d_in[0];
  const int*   bid    = (const int*)d_in[1];
  const unsigned char* mask = (const unsigned char*)d_in[2];
  const float* emb    = (const float*)d_in[3];
  const float* tokW1  = (const float*)d_in[4];
  const float* tokb1  = (const float*)d_in[5];
  const float* tokW2  = (const float*)d_in[6];
  const float* tokb2  = (const float*)d_in[7];
  const float* ln1g   = (const float*)d_in[8];
  const float* ln1b   = (const float*)d_in[9];
  const float* Wqkv   = (const float*)d_in[10];
  const float* bqkv   = (const float*)d_in[11];
  const float* Wout   = (const float*)d_in[12];
  const float* bout   = (const float*)d_in[13];
  const float* ln2g   = (const float*)d_in[14];
  const float* ln2b   = (const float*)d_in[15];
  const float* ffnW1  = (const float*)d_in[16];
  const float* ffnb1  = (const float*)d_in[17];
  const float* ffnW2  = (const float*)d_in[18];
  const float* ffnb2  = (const float*)d_in[19];
  const float* fng    = (const float*)d_in[20];
  const float* fnb    = (const float*)d_in[21];
  const float* pq     = (const float*)d_in[22];
  const float* pWq    = (const float*)d_in[23];
  const float* pWk    = (const float*)d_in[24];
  const float* pWv    = (const float*)d_in[25];
  const float* pbq    = (const float*)d_in[26];
  const float* pbk    = (const float*)d_in[27];
  const float* pbv    = (const float*)d_in[28];
  const float* pWo    = (const float*)d_in[29];
  const float* pbo    = (const float*)d_in[30];
  const float* plg    = (const float*)d_in[31];
  const float* plb    = (const float*)d_in[32];
  const float* clg    = (const float*)d_in[33];
  const float* clb    = (const float*)d_in[34];
  const float* ctxW   = (const float*)d_in[35];
  const float* ctxb   = (const float*)d_in[36];

  char* ws = (char*)d_ws;
  const size_t o_cs    = 0;                         // 4096*32*4   = 512 KB
  const size_t o_x     = o_cs + 524288;             // 4 MB
  const size_t o_y     = o_x + 4194304;             // 4 MB (LN out / xf)
  const size_t o_ao    = o_y + 4194304;             // 4 MB (attn out)
  const size_t o_big   = o_ao + 4194304;            // 16.8 MB (qkv | ffn-hidden | kh+vh)
  const size_t o_small = o_big + 16777216;          // qvec + pooled
  float* cs   = (float*)(ws + o_cs);
  float* x    = (float*)(ws + o_x);
  float* y    = (float*)(ws + o_y);
  float* ao   = (float*)(ws + o_ao);
  float* qkv  = (float*)(ws + o_big);
  float* hbuf = (float*)(ws + o_big);
  float* kh   = (float*)(ws + o_big);
  float* vh   = (float*)(ws + o_big + 4194304);
  float* qvec = (float*)(ws + o_small);
  float* po   = (float*)(ws + o_small + 2048);
  (void)in_sizes; (void)n_in; (void)out_size; (void)ws_size;

  k_embed<<<MTOK, 256, 0, stream>>>(feat, bid, emb, tokW1, tokb1, tokW2, tokb2, x, cs);

  for (int i = 0; i < NLY; i++) {
    k_ln<<<MTOK, 256, 0, stream>>>(x, ln1g + i * DMODEL, ln1b + i * DMODEL, y);
    k_gemm<0><<<dim3(12, 32), 256, 0, stream>>>(y, Wqkv + (size_t)i * DMODEL * 3 * DMODEL,
                                                bqkv + i * 3 * DMODEL, nullptr, qkv,
                                                MTOK, 3 * DMODEL, DMODEL);
    k_attn<<<dim3(BB * NH, LL / 128), 256, 0, stream>>>(qkv, cs, mask, ao);
    k_gemm<2><<<dim3(4, 32), 256, 0, stream>>>(ao, Wout + (size_t)i * DMODEL * DMODEL,
                                               bout + i * DMODEL, x, x,
                                               MTOK, DMODEL, DMODEL);
    k_ln<<<MTOK, 256, 0, stream>>>(x, ln2g + i * DMODEL, ln2b + i * DMODEL, y);
    k_gemm<1><<<dim3(16, 32), 256, 0, stream>>>(y, ffnW1 + (size_t)i * DMODEL * FFD,
                                                ffnb1 + i * FFD, nullptr, hbuf,
                                                MTOK, FFD, DMODEL);
    k_gemm<2><<<dim3(4, 32), 256, 0, stream>>>(hbuf, ffnW2 + (size_t)i * FFD * DMODEL,
                                               ffnb2 + i * DMODEL, x, x,
                                               MTOK, DMODEL, FFD);
  }

  // final LN -> y, then pooling K/V projections (WMMA), pooling attention, head
  k_ln<<<MTOK, 256, 0, stream>>>(x, fng, fnb, y);
  k_gemm<0><<<dim3(4, 32), 256, 0, stream>>>(y, pWk, pbk, nullptr, kh, MTOK, DMODEL, DMODEL);
  k_gemm<0><<<dim3(4, 32), 256, 0, stream>>>(y, pWv, pbv, nullptr, vh, MTOK, DMODEL, DMODEL);
  k_qproj<<<1, 256, 0, stream>>>(pq, pWq, pbq, qvec);
  k_pool_attn<<<BB * NH, 256, 0, stream>>>(qvec, kh, vh, mask, po);
  k_pool_final<<<BB, 256, 0, stream>>>(po, pWo, pbo, plg, plb, clg, clb, ctxW, ctxb,
                                       (float*)d_out);
}
